// TransformerBlock_24816321036591
// MI455X (gfx1250) — matrix operand order repaired and checked by reading, not compiled, not run on hardware
//
#include <hip/hip_runtime.h>
#include <math.h>

// ---------------------------------------------------------------------------
// Types for CDNA5 WMMA (wave32): D(16x16,f32) = A(16x32,bf16) x B(32x16,bf16) + C
// ---------------------------------------------------------------------------
typedef __attribute__((ext_vector_type(16))) __bf16 v16bf;
typedef __attribute__((ext_vector_type(8)))  float  v8f;
typedef __attribute__((ext_vector_type(4)))  unsigned int v4u;
typedef __attribute__((ext_vector_type(8)))  int v8i_;
typedef __attribute__((ext_vector_type(4)))  int v4i_;

#if __has_builtin(__builtin_amdgcn_tensor_load_to_lds) && \
    __has_builtin(__builtin_amdgcn_s_wait_tensorcnt)
#define HAVE_TDM 1
#else
#define HAVE_TDM 0
#endif

union FragBF {
  v16bf v;
  uint4 q[2];  // 32 bytes = 16 bf16
};

__device__ __forceinline__ v8f wmma_bf16(const FragBF& a, const FragBF& b, v8f c) {
  return __builtin_amdgcn_wmma_f32_16x16x32_bf16(
      /*neg_a=*/false, a.v, /*neg_b=*/false, b.v,
      /*c_mod=*/(short)0, c, /*reuse_a=*/false, /*reuse_b=*/false);
}

__device__ __forceinline__ unsigned short f2bf(float f) {
  unsigned int u = __float_as_uint(f);
  u += 0x7FFFu + ((u >> 16) & 1u);   // round-to-nearest-even
  return (unsigned short)(u >> 16);
}

#if HAVE_TDM
// ---------------------------------------------------------------------------
// TDM: issue a 2D tile load (tile_rows x tile_k bf16, row stride = K elems)
// from global memory into LDS at byte offset lds_off. D# per ISA §8.3-8.4.
// Group0: count=1 | lds_addr | global_addr(57b) | type=2.
// Group1: data_size=1(2B) | tensor_dim0/1 | tile_dim0/1 | tensor_dim0_stride.
// Groups 2/3 + extra group: zero (2D tensor). Tracked by TENSORcnt.
// clang-23 lane: 6-arg builtin (g0 v4u, g1 v8i, g2 v4i, g3 v4i, v8i, i32 cpol).
// ---------------------------------------------------------------------------
__device__ __forceinline__ void tdm_load_tile_2d(const unsigned short* gptr,
                                                 unsigned int lds_off,
                                                 unsigned int tile_k,
                                                 unsigned int tile_rows,
                                                 unsigned int stride_elems) {
  const unsigned long long ga = (unsigned long long)gptr;
  const unsigned int td0 = stride_elems;   // >= tile_k: never OOB in dim0
  const unsigned int td1 = 0x40000000u;    // huge: never OOB in dim1

  v4u g0;
  g0[0] = 1u;                                            // count=1, user D#
  g0[1] = lds_off;                                       // lds_addr (bytes)
  g0[2] = (unsigned int)(ga & 0xFFFFFFFFu);              // global_addr[31:0]
  g0[3] = (unsigned int)((ga >> 32) & 0x01FFFFFFu)       // global_addr[56:32]
          | (2u << 30);                                  // type=2 ("image")

  v8i_ g1;
  g1[0] = (int)(1u << 16);                               // data_size=1 (2B)
  g1[1] = (int)((td0 & 0xFFFFu) << 16);                  // tensor_dim0[15:0]
  g1[2] = (int)((td0 >> 16) | ((td1 & 0xFFFFu) << 16));  // td0[31:16]|td1[15:0]
  g1[3] = (int)((td1 >> 16) | (tile_k << 16));           // td1[31:16]|tile_dim0
  g1[4] = (int)(tile_rows & 0xFFFFu);                    // tile_dim1 (tile_dim2=0)
  g1[5] = (int)stride_elems;                             // tensor_dim0_stride[31:0]
  g1[6] = 0;                                             // stride[47:32]=0, dim1_stride lo
  g1[7] = 0;

  v4i_ z4 = {};
  v8i_ z8 = {};
  __builtin_amdgcn_tensor_load_to_lds(g0, g1, z4, z4, z8, 0);
}
#endif

// ---------------------------------------------------------------------------
// Elementwise f32 -> bf16 cast (weights; streaming, L2-resident afterwards)
// ---------------------------------------------------------------------------
__global__ __launch_bounds__(256) void cast_f32_bf16(const float* __restrict__ src,
                                                     unsigned short* __restrict__ dst,
                                                     int n) {
  int stride = gridDim.x * blockDim.x;
  for (int i = blockIdx.x * blockDim.x + threadIdx.x; i < n; i += stride)
    dst[i] = f2bf(src[i]);
}

// ---------------------------------------------------------------------------
// RMSNorm(row of 2048) * g  -> bf16.  One 256-thread block (8 waves) per row.
// ---------------------------------------------------------------------------
__global__ __launch_bounds__(256) void rmsnorm_cast(const float* __restrict__ x,
                                                    const float* __restrict__ g,
                                                    unsigned short* __restrict__ out) {
  const int D = 2048;
  const size_t row = blockIdx.x;
  const float* xr = x + row * D;
  float ss = 0.f;
  for (int i = threadIdx.x; i < D; i += 256) { float v = xr[i]; ss += v * v; }
  #pragma unroll
  for (int m = 1; m < 32; m <<= 1) ss += __shfl_xor(ss, m, 32);
  __shared__ float sred[8];
  if ((threadIdx.x & 31) == 0) sred[threadIdx.x >> 5] = ss;
  __syncthreads();
  float tot = 0.f;
  #pragma unroll
  for (int w = 0; w < 8; ++w) tot += sred[w];
  const float inv = rsqrtf(tot * (1.0f / 2048.0f) + 1e-5f);
  for (int i = threadIdx.x; i < D; i += 256)
    out[row * D + i] = f2bf(xr[i] * inv * g[i]);
}

// ---------------------------------------------------------------------------
// GEMM: C[M][N] = A[M][K](bf16,row-major) @ W[N][K]^T (bf16,row-major)
// Block = 128 threads = 4 waves (2x2), block tile 128x128, wave tile 64x64.
// TDM path: wave0 streams double-buffered 128x32 A/B tiles into LDS with
// tensor_load_to_lds (TENSORcnt), inner loop is ds_load_b128 -> v_wmma.
// MODE 0: bf16 out, *alpha           (Q / K projections)
// MODE 1: bf16 out, head-transposed  vt[b][h][d][s]      (V projection)
// MODE 2: f32 out = resid + acc      (O-proj into x1; FFN2 into d_out)
// MODE 3: bf16 out = gelu(acc)       (FFN1)
// ---------------------------------------------------------------------------
template <int MODE>
__global__ __launch_bounds__(128) void gemm_bf16(
    const unsigned short* __restrict__ A,
    const unsigned short* __restrict__ B,
    int M, int N, int K, float alpha,
    const float* __restrict__ resid,
    void* __restrict__ outp) {
  const int lane  = threadIdx.x & 31;
  const int wave  = threadIdx.x >> 5;
  const int lr    = lane & 15;
  const int lhalf = lane >> 4;
  const int wmRow = (wave >> 1) * 64;           // wave row offset in block tile
  const int wnRow = (wave & 1) * 64;            // wave col offset in block tile
  const int bm = blockIdx.y * 128 + wmRow;
  const int bn = blockIdx.x * 128 + wnRow;

  const int koffA = lhalf * 8;   // A lane k-chunks at {koffA, koffA+16}
  const int koffB = lhalf * 8;   // B lane k-chunks at {koffB, koffB+16}

  v8f acc[4][4];
  {
    v8f z = {};
    #pragma unroll
    for (int mt = 0; mt < 4; ++mt)
      #pragma unroll
      for (int nt = 0; nt < 4; ++nt) acc[mt][nt] = z;
  }

#if HAVE_TDM
  // [buf][A=0/B=1][row 0..127][k 0..31] ; 32 KB of the 320 KB WGP LDS pool.
  __shared__ __align__(16) unsigned short ldsAB[2][2][128][32];
  const unsigned int TILE_BYTES = 128u * 32u * 2u;           // 8 KB
  const unsigned short* Ablk = A + (size_t)(blockIdx.y * 128) * K;
  const unsigned short* Bblk = B + (size_t)(blockIdx.x * 128) * K;
  const int nk = K >> 5;

  if (wave == 0) {  // prologue: stage k-step 0 into buffer 0
    tdm_load_tile_2d(Ablk, 0u,              32u, 128u, (unsigned)K);
    tdm_load_tile_2d(Bblk, TILE_BYTES,      32u, 128u, (unsigned)K);
  }

  for (int ks = 0; ks < nk; ++ks) {
    const int buf = ks & 1;
    if (wave == 0) {
      if (ks + 1 < nk) {  // prefetch next k-step into the other buffer
        const unsigned int lb = (unsigned)(buf ^ 1) * 2u * TILE_BYTES;
        tdm_load_tile_2d(Ablk + (size_t)(ks + 1) * 32, lb,              32u, 128u, (unsigned)K);
        tdm_load_tile_2d(Bblk + (size_t)(ks + 1) * 32, lb + TILE_BYTES, 32u, 128u, (unsigned)K);
        __builtin_amdgcn_s_wait_tensorcnt(2);  // pairs complete in order
      } else {
        __builtin_amdgcn_s_wait_tensorcnt(0);
      }
    }
    __syncthreads();  // tile `buf` visible to all 4 waves

    FragBF a[4], b[4];
    #pragma unroll
    for (int mt = 0; mt < 4; ++mt) {
      const unsigned short* p = &ldsAB[buf][0][wmRow + mt * 16 + lr][koffA];
      a[mt].q[0] = *(const uint4*)(p);
      a[mt].q[1] = *(const uint4*)(p + 16);
    }
    #pragma unroll
    for (int nt = 0; nt < 4; ++nt) {
      const unsigned short* p = &ldsAB[buf][1][wnRow + nt * 16 + lr][koffB];
      b[nt].q[0] = *(const uint4*)(p);
      b[nt].q[1] = *(const uint4*)(p + 16);
    }
    #pragma unroll
    for (int mt = 0; mt < 4; ++mt)
      #pragma unroll
      for (int nt = 0; nt < 4; ++nt)
        acc[mt][nt] = wmma_bf16(a[mt], b[nt], acc[mt][nt]);

    __syncthreads();  // done reading before `buf` gets overwritten
  }
#else
  // Fallback: direct global-load fragments (known-good path).
  for (int k0 = 0; k0 < K; k0 += 32) {
    FragBF a[4], b[4];
    #pragma unroll
    for (int mt = 0; mt < 4; ++mt) {
      const unsigned short* p = A + (size_t)(bm + mt * 16 + lr) * K + k0 + koffA;
      a[mt].q[0] = *(const uint4*)(p);
      a[mt].q[1] = *(const uint4*)(p + 16);
    }
    #pragma unroll
    for (int nt = 0; nt < 4; ++nt) {
      const unsigned short* p = B + (size_t)(bn + nt * 16 + lr) * K + k0 + koffB;
      b[nt].q[0] = *(const uint4*)(p);
      b[nt].q[1] = *(const uint4*)(p + 16);
    }
    #pragma unroll
    for (int mt = 0; mt < 4; ++mt)
      #pragma unroll
      for (int nt = 0; nt < 4; ++nt)
        acc[mt][nt] = wmma_bf16(a[mt], b[nt], acc[mt][nt]);
  }
#endif

  // Epilogue: C/D layout => element r of v8f is row (r + lhalf*8), col lr.
  #pragma unroll
  for (int mt = 0; mt < 4; ++mt) {
    #pragma unroll
    for (int nt = 0; nt < 4; ++nt) {
      #pragma unroll
      for (int r = 0; r < 8; ++r) {
        const int m = bm + mt * 16 + r + lhalf * 8;
        const int n = bn + nt * 16 + lr;
        const float v = acc[mt][nt][r];
        if (MODE == 0) {
          ((unsigned short*)outp)[(size_t)m * N + n] = f2bf(v * alpha);
        } else if (MODE == 1) {
          const int b_ = m >> 11, s_ = m & 2047;   // token -> (batch, seq)
          const int h_ = n >> 7,  d_ = n & 127;    // col   -> (head, dim)
          const size_t idx = ((size_t)(b_ * 16 + h_) * 128 + d_) * 2048 + s_;
          ((unsigned short*)outp)[idx] = f2bf(v);
        } else if (MODE == 2) {
          const size_t idx = (size_t)m * N + n;
          ((float*)outp)[idx] = resid[idx] + v;
        } else {  // MODE == 3: exact GELU
          const float gl = 0.5f * v * (1.0f + erff(v * 0.70710678118654752f));
          ((unsigned short*)outp)[(size_t)m * N + n] = f2bf(gl);
        }
      }
    }
  }
}

// ---------------------------------------------------------------------------
// Flash attention, causal. One wave (32 threads) per (b, h, 16-row q tile).
// Q is pre-scaled by 1/sqrt(dk). K stored [token][h*128+d]; V stored
// transposed vt[b][h][d][s] so P@V B-fragments are contiguous loads.
// P converts C-layout -> A-layout through 1KB of LDS.
// ---------------------------------------------------------------------------
__global__ __launch_bounds__(32) void attention_kernel(
    const unsigned short* __restrict__ Qb,
    const unsigned short* __restrict__ Kb,
    const unsigned short* __restrict__ Vt,
    unsigned short* __restrict__ Ob) {
  const int S = 2048, D = 2048, DK = 128;
  const int lane  = threadIdx.x;
  const int lr    = lane & 15;
  const int lhalf = lane >> 4;
  const int q0 = blockIdx.x * 16;
  const int h  = blockIdx.y;
  const int b  = blockIdx.z;
  const size_t tokBase = (size_t)b * S;
  const float NINF = -__builtin_inff();

  __shared__ __align__(16) unsigned short Ps[16][32];

  // Q A-fragments: 4 chunks over dk=128
  FragBF qf[4];
  #pragma unroll
  for (int dc = 0; dc < 4; ++dc) {
    const unsigned short* p =
        Qb + (tokBase + q0 + lr) * D + h * DK + dc * 32 + lhalf * 8;
    qf[dc].q[0] = *(const uint4*)(p);
    qf[dc].q[1] = *(const uint4*)(p + 16);
  }

  v8f o[8];
  {
    v8f z = {};
    #pragma unroll
    for (int i = 0; i < 8; ++i) o[i] = z;
  }
  float mi[8], li[8];
  #pragma unroll
  for (int r = 0; r < 8; ++r) { mi[r] = NINF; li[r] = 0.0f; }

  for (int kk = 0; kk < q0 + 16; kk += 32) {
    // ---- scores: S[16 x 32] = Q(16x128) @ K_chunk^T ----
    v8f s0 = {}, s1 = {};
    #pragma unroll
    for (int dc = 0; dc < 4; ++dc) {
      FragBF bk0, bk1;
      const unsigned short* p0 =
          Kb + (tokBase + kk + lr) * D + h * DK + dc * 32 + lhalf * 8;
      const unsigned short* p1 =
          Kb + (tokBase + kk + 16 + lr) * D + h * DK + dc * 32 + lhalf * 8;
      bk0.q[0] = *(const uint4*)(p0); bk0.q[1] = *(const uint4*)(p0 + 16);
      bk1.q[0] = *(const uint4*)(p1); bk1.q[1] = *(const uint4*)(p1 + 16);
      s0 = wmma_bf16(qf[dc], bk0, s0);
      s1 = wmma_bf16(qf[dc], bk1, s1);
    }

    // ---- causal mask + online softmax (f32) ----
    #pragma unroll
    for (int r = 0; r < 8; ++r) {
      const int qrow = q0 + r + lhalf * 8;
      const int j0 = kk + lr;
      const int j1 = kk + 16 + lr;
      float e0 = (j0 <= qrow) ? s0[r] : NINF;
      float e1 = (j1 <= qrow) ? s1[r] : NINF;
      float mx = fmaxf(e0, e1);
      #pragma unroll
      for (int msk = 1; msk < 16; msk <<= 1)
        mx = fmaxf(mx, __shfl_xor(mx, msk, 32));
      const float mnew = fmaxf(mi[r], mx);
      const float p0 = __expf(e0 - mnew);
      const float p1 = __expf(e1 - mnew);
      const float scale = __expf(mi[r] - mnew);
      float ps = p0 + p1;
      #pragma unroll
      for (int msk = 1; msk < 16; msk <<= 1) ps += __shfl_xor(ps, msk, 32);
      li[r] = li[r] * scale + ps;
      mi[r] = mnew;
      #pragma unroll
      for (int dcc = 0; dcc < 8; ++dcc) o[dcc][r] *= scale;
      Ps[r + lhalf * 8][lr]      = f2bf(p0);
      Ps[r + lhalf * 8][16 + lr] = f2bf(p1);
    }
    __syncthreads();

    // ---- P (A-layout from LDS) @ V_chunk ----
    FragBF pf;
    pf.q[0] = *(const uint4*)&Ps[lr][lhalf * 8];
    pf.q[1] = *(const uint4*)&Ps[lr][lhalf * 8 + 16];
    #pragma unroll
    for (int dcc = 0; dcc < 8; ++dcc) {
      FragBF vf;
      const unsigned short* p =
          Vt + ((size_t)(b * 16 + h) * 128 + dcc * 16 + lr) * 2048 + kk + lhalf * 8;
      vf.q[0] = *(const uint4*)(p);
      vf.q[1] = *(const uint4*)(p + 16);
      o[dcc] = wmma_bf16(pf, vf, o[dcc]);
    }
    __syncthreads();
  }

  // ---- finalize: divide by row sums, store bf16 [token][h*128+d] ----
  float inv[8];
  #pragma unroll
  for (int r = 0; r < 8; ++r) inv[r] = 1.0f / li[r];
  #pragma unroll
  for (int dcc = 0; dcc < 8; ++dcc) {
    #pragma unroll
    for (int r = 0; r < 8; ++r) {
      const size_t idx =
          (tokBase + q0 + r + lhalf * 8) * (size_t)D + h * DK + dcc * 16 + lr;
      Ob[idx] = f2bf(o[dcc][r] * inv[r]);
    }
  }
}

// ---------------------------------------------------------------------------
// Host orchestration
// ---------------------------------------------------------------------------
extern "C" void kernel_launch(void* const* d_in, const int* in_sizes, int n_in,
                              void* d_out, int out_size, void* d_ws, size_t ws_size,
                              hipStream_t stream) {
  (void)in_sizes; (void)n_in; (void)out_size; (void)ws_size;

  const float* x  = (const float*)d_in[0];
  const float* wq = (const float*)d_in[1];
  const float* wk = (const float*)d_in[2];
  const float* wv = (const float*)d_in[3];
  const float* wo = (const float*)d_in[4];
  const float* w1 = (const float*)d_in[5];
  const float* w2 = (const float*)d_in[6];
  const float* g1 = (const float*)d_in[7];
  const float* g2 = (const float*)d_in[8];

  const int D = 2048, FF = 8192, T = 4096;  // T = B*S tokens

  char* ws = (char*)d_ws;
  size_t off = 0;
  auto alloc = [&](size_t bytes) -> void* {
    void* p = ws + off;
    off = (off + bytes + 255) & ~(size_t)255;
    return p;
  };

  unsigned short* wq_bf = (unsigned short*)alloc((size_t)D * D * 2);
  unsigned short* wk_bf = (unsigned short*)alloc((size_t)D * D * 2);
  unsigned short* wv_bf = (unsigned short*)alloc((size_t)D * D * 2);
  unsigned short* wo_bf = (unsigned short*)alloc((size_t)D * D * 2);
  unsigned short* w1_bf = (unsigned short*)alloc((size_t)FF * D * 2);
  unsigned short* w2_bf = (unsigned short*)alloc((size_t)D * FF * 2);
  unsigned short* xn    = (unsigned short*)alloc((size_t)T * D * 2);
  unsigned short* qb    = (unsigned short*)alloc((size_t)T * D * 2);
  unsigned short* kb    = (unsigned short*)alloc((size_t)T * D * 2);
  unsigned short* vt    = (unsigned short*)alloc((size_t)T * D * 2);
  unsigned short* ab    = (unsigned short*)alloc((size_t)T * D * 2);
  float*          x1    = (float*)alloc((size_t)T * D * 4);
  unsigned short* x1n   = (unsigned short*)alloc((size_t)T * D * 2);
  unsigned short* hb    = (unsigned short*)alloc((size_t)T * FF * 2);

  // 1) weight casts to bf16
  const int nDD = D * D, nFD = FF * D;
  const int cgrid = 8192;
  cast_f32_bf16<<<cgrid, 256, 0, stream>>>(wq, wq_bf, nDD);
  cast_f32_bf16<<<cgrid, 256, 0, stream>>>(wk, wk_bf, nDD);
  cast_f32_bf16<<<cgrid, 256, 0, stream>>>(wv, wv_bf, nDD);
  cast_f32_bf16<<<cgrid, 256, 0, stream>>>(wo, wo_bf, nDD);
  cast_f32_bf16<<<cgrid, 256, 0, stream>>>(w1, w1_bf, nFD);
  cast_f32_bf16<<<cgrid, 256, 0, stream>>>(w2, w2_bf, nFD);

  // 2) RMSNorm 1
  rmsnorm_cast<<<T, 256, 0, stream>>>(x, g1, xn);

  // 3) Q, K, V projections (Q pre-scaled by dk^-0.5; V head-transposed)
  const dim3 blk(128);
  const dim3 gD(D / 128, T / 128);
  const dim3 gF(FF / 128, T / 128);
  gemm_bf16<0><<<gD, blk, 0, stream>>>(xn, wq_bf, T, D, D, 0.08838834764831845f, nullptr, qb);
  gemm_bf16<0><<<gD, blk, 0, stream>>>(xn, wk_bf, T, D, D, 1.0f, nullptr, kb);
  gemm_bf16<1><<<gD, blk, 0, stream>>>(xn, wv_bf, T, D, D, 1.0f, nullptr, vt);

  // 4) causal flash attention
  attention_kernel<<<dim3(2048 / 16, 16, 2), dim3(32), 0, stream>>>(qb, kb, vt, ab);

  // 5) O projection + residual -> x1 (f32)
  gemm_bf16<2><<<gD, blk, 0, stream>>>(ab, wo_bf, T, D, D, 1.0f, x, x1);

  // 6) RMSNorm 2
  rmsnorm_cast<<<T, 256, 0, stream>>>(x1, g2, x1n);

  // 7) FFN
  gemm_bf16<3><<<gF, blk, 0, stream>>>(x1n, w1_bf, T, FF, D, 1.0f, nullptr, hb);
  gemm_bf16<2><<<gD, blk, 0, stream>>>(hb, w2_bf, T, D, FF, 1.0f, x1, (float*)d_out);
}
